// WindowAttention_34935263985945
// MI455X (gfx1250) — compile-verified, matrix operand order checked by reading, not run on hardware
//
#include <hip/hip_runtime.h>
#include <hip/hip_bf16.h>
#include <math.h>

typedef __attribute__((ext_vector_type(16))) _Float16 v16h;
typedef __attribute__((ext_vector_type(8)))  float    v8f;

constexpr int CCH    = 192;
constexpr int NHEAD  = 6;
constexpr int NTOK   = 64;        // tokens per window (4x4x4)
constexpr int DHW    = 131072;    // 32*64*64
constexpr int NWIN   = 2048;      // windows per batch (8*16*16)
constexpr int TOKENS = 262144;    // 2*DHW

__device__ __forceinline__ v8f wmma16(v16h a, v16h b, v8f c) {
  return __builtin_amdgcn_wmma_f32_16x16x32_f16(false, a, false, b, (short)0, c,
                                                false, false);
}

// Load a 16x32 f16 A/B fragment from an LDS matrix stored [row][k] with row
// stride ld (halves). For A: row = M. For B (K x N, B[k][n] = src[n][k]): row = N.
// Layout per CDNA5 ISA 7.12.2: lane<16 holds row (lane&15), K = k0..k0+7 in
// halves 0..7 and K = k0+16..k0+23 in halves 8..15; lane>=16 gets K offset +8.
__device__ __forceinline__ v16h ldfrag(const _Float16* base, int ld, int r0,
                                       int k0, int lane) {
  const _Float16* p = base + (r0 + (lane & 15)) * ld + k0 + ((lane >> 4) << 3);
  v16h f;
#pragma unroll
  for (int i = 0; i < 8; ++i) f[i] = p[i];
#pragma unroll
  for (int i = 0; i < 8; ++i) f[8 + i] = p[16 + i];
  return f;
}

__device__ __forceinline__ v8f zero8() {
  v8f z = {0.f, 0.f, 0.f, 0.f, 0.f, 0.f, 0.f, 0.f};
  return z;
}

// ---------------- Stage 0: one-time weight prep (f32 -> f16, reorder) --------
// wqkv: [head][96][192] f16 (rows j: 0..31=q, 32..63=k, 64..95=v for that head)
// bqkv: [head][96] f32; wproj: [192][192] f16; wfc1: [768][192] f16;
// wfc2: [192][768] f16.
__global__ __launch_bounds__(256) void k_wprep(
    const float* __restrict__ qkv_w, const float* __restrict__ qkv_b,
    const float* __restrict__ proj_w, const float* __restrict__ fc1_w,
    const float* __restrict__ fc2_w, _Float16* __restrict__ wqkv,
    float* __restrict__ bqkv, _Float16* __restrict__ wproj,
    _Float16* __restrict__ wfc1, _Float16* __restrict__ wfc2) {
  const int stride = gridDim.x * 256;
  const int t0 = blockIdx.x * 256 + threadIdx.x;
  for (int idx = t0; idx < NHEAD * 96 * CCH; idx += stride) {
    int k = idx % CCH;
    int j = (idx / CCH) % 96;
    int h = idx / (96 * CCH);
    int srow = (j < 32) ? (h * 32 + j)
                        : ((j < 64) ? (CCH + h * 32 + j - 32)
                                    : (2 * CCH + h * 32 + j - 64));
    wqkv[idx] = (_Float16)qkv_w[srow * CCH + k];
  }
  for (int idx = t0; idx < NHEAD * 96; idx += stride) {
    int j = idx % 96, h = idx / 96;
    int srow = (j < 32) ? (h * 32 + j)
                        : ((j < 64) ? (CCH + h * 32 + j - 32)
                                    : (2 * CCH + h * 32 + j - 64));
    bqkv[idx] = qkv_b[srow];
  }
  for (int idx = t0; idx < CCH * CCH; idx += stride)
    wproj[idx] = (_Float16)proj_w[idx];
  for (int idx = t0; idx < 768 * CCH; idx += stride)
    wfc1[idx] = (_Float16)fc1_w[idx];
  for (int idx = t0; idx < CCH * 768; idx += stride)
    wfc2[idx] = (_Float16)fc2_w[idx];
}

// ---------------- Stage 1: LN1 + cyclic shift + window partition -------------
__global__ __launch_bounds__(256) void k_ln1(const float* __restrict__ x,
                                             const float* __restrict__ g1,
                                             const float* __restrict__ b1,
                                             _Float16* __restrict__ xw,
                                             float* __restrict__ xt) {
  __shared__ float T[192 * 132];
  __shared__ float MU[128];
  __shared__ float RS[128];
  const int tid = threadIdx.x;
  const int b = blockIdx.x >> 10;
  const int s0 = (blockIdx.x & 1023) * 128;
  const float* xb = x + (size_t)b * CCH * DHW;

  for (int idx = tid; idx < 192 * 128; idx += 256) {
    int c = idx >> 7, s = idx & 127;
    T[c * 132 + s] = xb[(size_t)c * DHW + s0 + s];
  }
  __syncthreads();
  if (tid < 128) {
    float m = 0.f;
    for (int c = 0; c < CCH; ++c) m += T[c * 132 + tid];
    m *= (1.f / 192.f);
    float v = 0.f;
    for (int c = 0; c < CCH; ++c) {
      float d = T[c * 132 + tid] - m;
      v += d * d;
    }
    v *= (1.f / 192.f);
    MU[tid] = m;
    RS[tid] = rsqrtf(v + 1e-5f);
  }
  __syncthreads();
  // fp32 shortcut, token-major (b,d,h,w,c)
  for (int idx = tid; idx < 128 * 192; idx += 256) {
    int s = idx / 192, c = idx % 192;
    xt[((size_t)b * DHW + s0 + s) * CCH + c] = T[c * 132 + s];
  }
  // LN + shift + partition -> xw
  for (int idx = tid; idx < 128 * 192; idx += 256) {
    int s = idx / 192, c = idx % 192;
    int sp = s0 + s;
    int d = sp >> 12, h = (sp >> 6) & 63, w = sp & 63;
    int gd = (d + 30) & 31, gh = (h + 62) & 63, gw = (w + 62) & 63;  // roll -2
    int win = ((gd >> 2) * 16 + (gh >> 2)) * 16 + (gw >> 2);
    int loc = ((gd & 3) * 4 + (gh & 3)) * 4 + (gw & 3);
    size_t row = ((size_t)b * NWIN + win) * NTOK + loc;
    float v = (T[c * 132 + s] - MU[s]) * RS[s] * g1[c] + b1[c];
    xw[row * CCH + c] = (_Float16)v;
  }
}

// ---------------- Stage 2: fused qkv + attention + proj + residual -----------
constexpr int A_XS  = 0;                 // 64 x 200 f16 = 25600
constexpr int A_WS  = 25600;             // 96 x 200 f16 = 38400
constexpr int A_QK  = 64000;             // 64 x 72 f16  = 9216
constexpr int A_VT  = 73216;             // 32 x 72 f16  = 4608
constexpr int A_PW  = 0;                 // overlay: 192 x 200 f16 = 76800 <= 77824
constexpr int A_PS  = 77824;             // 64 x 72 f16  = 9216
constexpr int A_OS  = 87040;             // 64 x 200 f16 = 25600
constexpr int A_RPB = 112640;            // 343*6 f32    = 8232
constexpr int A_LBL = 120872;            // 64 i32       = 256
constexpr int A_SMEM = 121216;

__global__ __launch_bounds__(128) void k_attn(
    const _Float16* __restrict__ xw, const _Float16* __restrict__ wqkv,
    const float* __restrict__ bqkv, const float* __restrict__ rpb,
    const _Float16* __restrict__ wproj, const float* __restrict__ proj_b,
    float* __restrict__ xt) {
  __shared__ __align__(16) char sm[A_SMEM];
  _Float16* Xs = (_Float16*)(sm + A_XS);
  _Float16* Ws = (_Float16*)(sm + A_WS);
  _Float16* Qk = (_Float16*)(sm + A_QK);
  _Float16* Vt = (_Float16*)(sm + A_VT);
  _Float16* Pw = (_Float16*)(sm + A_PW);
  _Float16* Ps = (_Float16*)(sm + A_PS);
  _Float16* Os = (_Float16*)(sm + A_OS);
  float*    Rp = (float*)(sm + A_RPB);
  int*      Lb = (int*)(sm + A_LBL);

  const int tid = threadIdx.x, lane = tid & 31, wave = tid >> 5;
  const int hh = lane >> 4;
  const int bwin = blockIdx.x;
  const int b = bwin >> 11, win = bwin & 2047;
  const int wz = win >> 8, wy = (win >> 4) & 15, wx = win & 15;
  const int m0 = wave * 16;

  // stage window tokens with 128-bit copies (row = 24 uint4, LDS row = 400B)
  const uint4* xwin4 = (const uint4*)(xw + (size_t)bwin * NTOK * CCH);
  for (int idx = tid; idx < NTOK * 24; idx += 128) {
    int r = idx / 24, c4 = idx % 24;
    *(uint4*)(sm + A_XS + r * 400 + c4 * 16) = xwin4[idx];
  }
  for (int idx = tid; idx < 343 * NHEAD; idx += 128) Rp[idx] = rpb[idx];
  if (tid < 64) {
    int ld = tid >> 4, lh = (tid >> 2) & 3, lw = tid & 3;
    int gd = wz * 4 + ld, gh = wy * 4 + lh, gw = wx * 4 + lw;
    int rd = (gd < 28) ? 0 : ((gd < 30) ? 1 : 2);
    int rh = (gh < 60) ? 0 : ((gh < 62) ? 1 : 2);
    int rw = (gw < 60) ? 0 : ((gw < 62) ? 1 : 2);
    Lb[tid] = (rd * 3 + rh) * 3 + rw;
  }
  __syncthreads();

  const float scale = 0.17677669529663687f;  // 32^-0.5

  for (int h = 0; h < NHEAD; ++h) {
    // per-head q,k,v weight panel (96 x 192 f16, contiguous) -> LDS
    const uint4* wsrc = (const uint4*)(wqkv + (size_t)h * 96 * CCH);
    for (int idx = tid; idx < 96 * 24; idx += 128) {
      int j = idx / 24, c4 = idx % 24;
      *(uint4*)(sm + A_WS + j * 400 + c4 * 16) = wsrc[idx];
    }
    __syncthreads();

    // qkv GEMM: 16-row strip x 96 cols, K=192
    v8f acc[6];
    for (int t = 0; t < 6; ++t) acc[t] = zero8();
    for (int kc = 0; kc < 6; ++kc) {
      v16h a = ldfrag(Xs, 200, m0, kc * 32, lane);
      for (int nt = 0; nt < 6; ++nt) {
        v16h bb = ldfrag(Ws, 200, nt * 16, kc * 32, lane);
        acc[nt] = wmma16(a, bb, acc[nt]);
      }
    }
    for (int nt = 0; nt < 6; ++nt) {
      int n = nt * 16 + (lane & 15);
      float bb = bqkv[h * 96 + n];
      for (int i = 0; i < 8; ++i) {
        int m = m0 + i + 8 * hh;
        float v = acc[nt][i] + bb;
        if (n < 64) Qk[m * 72 + n] = (_Float16)v;        // q cols 0..31, k 32..63
        else        Vt[(n - 64) * 72 + m] = (_Float16)v;  // v transposed [col][tok]
      }
    }
    __syncthreads();

    // S = q @ k^T (K = 32)
    v8f s[4];
    for (int t = 0; t < 4; ++t) s[t] = zero8();
    {
      v16h aq = ldfrag(Qk, 72, m0, 0, lane);
      for (int nt = 0; nt < 4; ++nt) {
        v16h bk = ldfrag(Qk + 32, 72, nt * 16, 0, lane);
        s[nt] = wmma16(aq, bk, s[nt]);
      }
    }
    // scale + rel-pos bias + shifted-window mask
    for (int nt = 0; nt < 4; ++nt) {
      int n2 = nt * 16 + (lane & 15);
      int dn = n2 >> 4, hn = (n2 >> 2) & 3, wn = n2 & 3;
      int ln2 = Lb[n2];
      for (int i = 0; i < 8; ++i) {
        int m = m0 + i + 8 * hh;
        int dm = m >> 4, hm = (m >> 2) & 3, wm = m & 3;
        int ridx = (dm - dn + 3) * 49 + (hm - hn + 3) * 7 + (wm - wn + 3);
        float bias = Rp[ridx * NHEAD + h];
        float msk = (Lb[m] != ln2) ? -100.f : 0.f;
        s[nt][i] = s[nt][i] * scale + bias + msk;
      }
    }
    // softmax per row (row lives in fixed vgpr across 16 lanes of a half)
    for (int i = 0; i < 8; ++i) {
      float mx = fmaxf(fmaxf(s[0][i], s[1][i]), fmaxf(s[2][i], s[3][i]));
      for (int o = 8; o >= 1; o >>= 1) mx = fmaxf(mx, __shfl_xor(mx, o, 32));
      float sum = 0.f;
      for (int t = 0; t < 4; ++t) {
        s[t][i] = __expf(s[t][i] - mx);
        sum += s[t][i];
      }
      for (int o = 8; o >= 1; o >>= 1) sum += __shfl_xor(sum, o, 32);
      float inv = 1.f / sum;
      for (int t = 0; t < 4; ++t) s[t][i] *= inv;
    }
    // stage P strip (own rows only -> no barrier needed)
    for (int nt = 0; nt < 4; ++nt) {
      int n2 = nt * 16 + (lane & 15);
      for (int i = 0; i < 8; ++i) {
        int m = m0 + i + 8 * hh;
        Ps[m * 72 + n2] = (_Float16)s[nt][i];
      }
    }
    // O = P @ V (K = 64, N = 32)
    v8f o[2];
    o[0] = zero8();
    o[1] = zero8();
    for (int kc = 0; kc < 2; ++kc) {
      v16h ap = ldfrag(Ps, 72, m0, kc * 32, lane);
      for (int nt = 0; nt < 2; ++nt) {
        v16h bv = ldfrag(Vt, 72, nt * 16, kc * 32, lane);
        o[nt] = wmma16(ap, bv, o[nt]);
      }
    }
    for (int nt = 0; nt < 2; ++nt) {
      int n = nt * 16 + (lane & 15);
      for (int i = 0; i < 8; ++i) {
        int m = m0 + i + 8 * hh;
        Os[m * 200 + h * 32 + n] = (_Float16)o[nt][i];
      }
    }
    __syncthreads();
  }

  // proj: O(64x192) @ proj_w^T + bias + window-reverse + un-shift + residual
  {
    const uint4* ps4 = (const uint4*)wproj;
    for (int idx = tid; idx < CCH * 24; idx += 128) {
      int r = idx / 24, c4 = idx % 24;
      *(uint4*)(sm + A_PW + r * 400 + c4 * 16) = ps4[idx];
    }
  }
  __syncthreads();
  v8f pa[12];
  for (int t = 0; t < 12; ++t) pa[t] = zero8();
  for (int kc = 0; kc < 6; ++kc) {
    v16h a = ldfrag(Os, 200, m0, kc * 32, lane);
    for (int nt = 0; nt < 12; ++nt) {
      v16h bb = ldfrag(Pw, 200, nt * 16, kc * 32, lane);
      pa[nt] = wmma16(a, bb, pa[nt]);
    }
  }
  for (int nt = 0; nt < 12; ++nt) {
    int n = nt * 16 + (lane & 15);
    float pb = proj_b[n];
    for (int i = 0; i < 8; ++i) {
      int m = m0 + i + 8 * hh;
      int gd = wz * 4 + (m >> 4);
      int gh = wy * 4 + ((m >> 2) & 3);
      int gw = wx * 4 + (m & 3);
      int od = (gd + 2) & 31, oh = (gh + 2) & 63, ow = (gw + 2) & 63;  // roll +2
      size_t tok = (size_t)b * DHW + (size_t)od * 4096 + oh * 64 + ow;
      float* p = xt + tok * CCH + n;
      *p = *p + pa[nt][i] + pb;
    }
  }
}

// ---------------- Stage 3: fused LN2 + MLP + residual ------------------------
constexpr int M_H2  = 0;          // 128 x 200 f16 = 51200
constexpr int M_W1  = 51200;      // 64 x 200 f16  = 25600
constexpr int M_HB  = 76800;      // 128 x 72 f16  = 18432
constexpr int M_W2  = 95232;      // 192 x 72 f16  = 27648
constexpr int M_MU  = 122880;     // 128 f32
constexpr int M_RS  = 123392;     // 128 f32
constexpr int M_SMEM = 123904;

__global__ __launch_bounds__(256) void k_mlp(float* __restrict__ xc,
                                             const float* __restrict__ g2,
                                             const float* __restrict__ b2,
                                             const _Float16* __restrict__ w1,
                                             const float* __restrict__ b1f,
                                             const _Float16* __restrict__ w2,
                                             const float* __restrict__ b2f) {
  __shared__ __align__(16) char sm[M_SMEM];
  _Float16* H2 = (_Float16*)(sm + M_H2);
  _Float16* HB = (_Float16*)(sm + M_HB);
  _Float16* W1 = (_Float16*)(sm + M_W1);
  _Float16* W2 = (_Float16*)(sm + M_W2);
  float* MU = (float*)(sm + M_MU);
  float* RS = (float*)(sm + M_RS);

  const int tid = threadIdx.x, lane = tid & 31, wave = tid >> 5;
  const int hh = lane >> 4;
  const size_t r0 = (size_t)blockIdx.x * 128;
  const int m0 = wave * 16;

  if (tid < 128) {
    const float* p = xc + (r0 + tid) * CCH;
    float m = 0.f;
    for (int c = 0; c < CCH; ++c) m += p[c];
    m *= (1.f / 192.f);
    float v = 0.f;
    for (int c = 0; c < CCH; ++c) {
      float d = p[c] - m;
      v += d * d;
    }
    v *= (1.f / 192.f);
    MU[tid] = m;
    RS[tid] = rsqrtf(v + 1e-5f);
  }
  __syncthreads();
  for (int idx = tid; idx < 128 * CCH; idx += 256) {
    int r = idx / CCH, c = idx % CCH;
    H2[r * 200 + c] =
        (_Float16)((xc[(r0 + r) * CCH + c] - MU[r]) * RS[r] * g2[c] + b2[c]);
  }
  __syncthreads();

  v8f a2[12];
  for (int t = 0; t < 12; ++t) a2[t] = zero8();

  for (int jc = 0; jc < 12; ++jc) {  // hidden dim 768 in chunks of 64
    __syncthreads();
    {
      const uint4* w1s = (const uint4*)(w1 + (size_t)jc * 64 * CCH);
      for (int idx = tid; idx < 64 * 24; idx += 256) {
        int j = idx / 24, c4 = idx % 24;
        *(uint4*)(sm + M_W1 + j * 400 + c4 * 16) = w1s[idx];
      }
    }
    __syncthreads();
    // fc1 chunk: strip(16) x 64, K=192
    v8f a1[4];
    for (int t = 0; t < 4; ++t) a1[t] = zero8();
    for (int kc = 0; kc < 6; ++kc) {
      v16h a = ldfrag(H2, 200, m0, kc * 32, lane);
      for (int nt = 0; nt < 4; ++nt) {
        v16h bb = ldfrag(W1, 200, nt * 16, kc * 32, lane);
        a1[nt] = wmma16(a, bb, a1[nt]);
      }
    }
    // exact GELU, stage to LDS (own rows)
    for (int nt = 0; nt < 4; ++nt) {
      int j = nt * 16 + (lane & 15);
      float bj = b1f[jc * 64 + j];
      for (int i = 0; i < 8; ++i) {
        float v = a1[nt][i] + bj;
        v = 0.5f * v * (1.f + erff(v * 0.70710678118654752f));
        int m = m0 + i + 8 * hh;
        HB[m * 72 + j] = (_Float16)v;
      }
    }
    __syncthreads();
    {
      // w2 rows: 8 uint4 per row chunk of 64 halves (row stride 768 halves)
      for (int idx = tid; idx < 192 * 8; idx += 256) {
        int n = idx >> 3, k4 = idx & 7;
        *(uint4*)(sm + M_W2 + n * 144 + k4 * 16) =
            *(const uint4*)(w2 + (size_t)n * 768 + jc * 64 + k4 * 8);
      }
    }
    __syncthreads();
    // fc2 accumulate: strip(16) x 192, K=64
    for (int kc = 0; kc < 2; ++kc) {
      v16h a = ldfrag(HB, 72, m0, kc * 32, lane);
      for (int nt = 0; nt < 12; ++nt) {
        v16h bb = ldfrag(W2, 72, nt * 16, kc * 32, lane);
        a2[nt] = wmma16(a, bb, a2[nt]);
      }
    }
  }
  // residual write (in place; each element touched by exactly one lane)
  for (int nt = 0; nt < 12; ++nt) {
    int n = nt * 16 + (lane & 15);
    float bn = b2f[n];
    for (int i = 0; i < 8; ++i) {
      int m = m0 + i + 8 * hh;
      size_t a = (r0 + m) * CCH + n;
      xc[a] = xc[a] + a2[nt][i] + bn;
    }
  }
}

// ---------------- Stage 4: transpose token-major -> (B,C,D,H,W) --------------
__global__ __launch_bounds__(256) void k_out(const float* __restrict__ xc,
                                             float* __restrict__ out) {
  __shared__ float T[192 * 132];
  const int tid = threadIdx.x;
  const int b = blockIdx.x >> 10;
  const int s0 = (blockIdx.x & 1023) * 128;
  for (int idx = tid; idx < 128 * 192; idx += 256) {
    int s = idx / 192, c = idx % 192;
    T[c * 132 + s] = xc[((size_t)b * DHW + s0 + s) * CCH + c];
  }
  __syncthreads();
  for (int idx = tid; idx < 192 * 128; idx += 256) {
    int c = idx >> 7, s = idx & 127;
    out[(size_t)b * CCH * DHW + (size_t)c * DHW + s0 + s] = T[c * 132 + s];
  }
}

extern "C" void kernel_launch(void* const* d_in, const int* in_sizes, int n_in,
                              void* d_out, int out_size, void* d_ws,
                              size_t ws_size, hipStream_t stream) {
  const float* x      = (const float*)d_in[0];
  const float* g1     = (const float*)d_in[1];
  const float* b1     = (const float*)d_in[2];
  const float* qkv_w  = (const float*)d_in[3];
  const float* qkv_b  = (const float*)d_in[4];
  const float* rpb    = (const float*)d_in[5];
  const float* proj_w = (const float*)d_in[6];
  const float* proj_b = (const float*)d_in[7];
  const float* g2     = (const float*)d_in[8];
  const float* b2     = (const float*)d_in[9];
  const float* fc1_w  = (const float*)d_in[10];
  const float* fc1_b  = (const float*)d_in[11];
  const float* fc2_w  = (const float*)d_in[12];
  const float* fc2_b  = (const float*)d_in[13];
  float* out = (float*)d_out;

  // workspace layout (all 16B aligned)
  char* ws = (char*)d_ws;
  size_t off = 0;
  _Float16* xw = (_Float16*)(ws + off); off += (size_t)TOKENS * CCH * 2;  // 100.7 MB
  float*    xt = (float*)(ws + off);    off += (size_t)TOKENS * CCH * 4;  // 201.3 MB
  _Float16* wqkv  = (_Float16*)(ws + off); off += (size_t)NHEAD * 96 * CCH * 2;
  float*    bqkv  = (float*)(ws + off);    off += (size_t)NHEAD * 96 * 4;
  _Float16* wproj = (_Float16*)(ws + off); off += (size_t)CCH * CCH * 2;
  _Float16* wfc1  = (_Float16*)(ws + off); off += (size_t)768 * CCH * 2;
  _Float16* wfc2  = (_Float16*)(ws + off); off += (size_t)CCH * 768 * 2;

  k_wprep<<<dim3(256), dim3(256), 0, stream>>>(qkv_w, qkv_b, proj_w, fc1_w,
                                               fc2_w, wqkv, bqkv, wproj, wfc1,
                                               wfc2);
  k_ln1<<<dim3(2 * 1024), dim3(256), 0, stream>>>(x, g1, b1, xw, xt);
  k_attn<<<dim3(2 * NWIN), dim3(128), 0, stream>>>(xw, wqkv, bqkv, rpb, wproj,
                                                   proj_b, xt);
  k_mlp<<<dim3(TOKENS / 128), dim3(256), 0, stream>>>(xt, g2, b2, wfc1, fc1_b,
                                                      wfc2, fc2_b);
  k_out<<<dim3(2 * 1024), dim3(256), 0, stream>>>(xt, out);
  (void)in_sizes; (void)n_in; (void)out_size; (void)ws_size;
}
